// WaveNOMinimal_75548474737143
// MI455X (gfx1250) — matrix operand order repaired and checked by reading, not compiled, not run on hardware
//
#include <hip/hip_runtime.h>

// ---------------------------------------------------------------------------
// WaveNO forward for MI455X (gfx1250, wave32, WMMA bf16 16x16x32 + TDM).
// One workgroup per (b, t): residual stream (256x64 f32) lives in LDS,
// bf16 weights staged to LDS via tensor_load_to_lds (TDM), all GEMMs on
// v_wmma_f32_16x16x32_bf16, row reductions via v_permlane16.
// ---------------------------------------------------------------------------

#define B_    8
#define NT_   256
#define NX_   256
#define KSEG  32
#define HD_   64
#define NH_   4
#define DH_   16
#define NL_   2
#define NF_   8
#define FDIM_ 17
#define FF_   256
#define BN_   (B_ * NT_)

typedef __bf16 bf16;
typedef unsigned int u32;
typedef __attribute__((ext_vector_type(16))) __bf16 v16bf;
typedef __attribute__((ext_vector_type(8)))  float  v8f;
typedef __attribute__((ext_vector_type(4)))  u32    v4u;
typedef __attribute__((ext_vector_type(8)))  int    v8i;
typedef __attribute__((ext_vector_type(4)))  int    v4i;

struct LayerP {
  const float *W1, *W2, *Wk, *Wo, *Wq, *Wv;
  const float *b1, *b2, *bk, *bo, *bq, *bv;
  const float *ln1b, *ln1g, *ln2b, *ln2g;
};
struct DevP {
  const float *xs, *ks; const int *pm; const float *tco, *xco;
  const float *bias_scale, *headW1, *headW2, *headB1, *headB2;
  const float *qryW1, *qryW2, *qryB1, *qryB2;
  const float *segW1, *segW2, *segB1, *segB2;
  LayerP L[NL_];
};

// ---- workspace layout (bf16 elements) -------------------------------------
// [ KV: (l,b) -> [Kh 32x64 | V 32x64] ] [ per-layer: [Wq|Wo][W1|W2] ] [ headW1 ]
constexpr int KV_PER_LB  = 2 * KSEG * HD_;              // 4096
constexpr int KV_ELEMS   = NL_ * B_ * KV_PER_LB;        // 65536
constexpr int LAY_OFF    = KV_ELEMS;
constexpr int LAY_STRIDE = 2 * HD_ * HD_ + 2 * HD_ * FF_;  // 8192 + 32768
constexpr int HEAD_OFF   = LAY_OFF + NL_ * LAY_STRIDE;  // 147456
// total 151552 bf16 elems = 303104 bytes of d_ws

// ---- dynamic LDS carve (bytes) --------------------------------------------
constexpr int OFF_SX   = 0;                                   // 256x64 f32
constexpr int OFF_BIAS = OFF_SX   + NX_ * HD_ * 4;            // 256x32 f32
constexpr int OFF_W    = OFF_BIAS + NX_ * KSEG * 4;           // 64KB bf16 weights
constexpr int OFF_KV   = OFF_W    + 65536;                    // [Kh|V] bf16
constexpr int OFF_SCR  = OFF_KV   + KV_PER_LB * 2;            // 8 waves x 16x64 f32
constexpr int OFF_SCR2 = OFF_SCR  + 8 * 16 * 64 * 4;          // 8 waves x 16x32 f32
constexpr int OFF_XC   = OFF_SCR2 + 8 * 16 * 32 * 4;          // 256 f32 x-coords
constexpr int OFF_KA   = OFF_XC   + NX_ * 4;                  // xm/speed/mask
constexpr int SMEM_BYTES = OFF_KA + 512;                      // ~218 KB < 320 KB/WGP

// ---- TDM: 1-D bf16 copy global -> LDS (D# per CDNA5 ISA) --
#define HAVE_TDM (__has_builtin(__builtin_amdgcn_tensor_load_to_lds))

static __device__ inline void tdm_load_1d(u32 lds_off, const void* gsrc, u32 nelem) {
#if HAVE_TDM
  unsigned long long ga = (unsigned long long)gsrc;
  v4u g0;
  g0[0] = 1u;                                   // count=1, user mode
  g0[1] = lds_off;                              // lds_addr (bytes)
  g0[2] = (u32)ga;                              // global_addr[31:0]
  g0[3] = (u32)((ga >> 32) & 0x01FFFFFFu) | (2u << 30);  // addr[56:32] | type=2
  v8i g1;
  g1[0] = (int)(1u << 16);                      // data_size=1 -> 2 bytes/elem
  g1[1] = (int)((nelem & 0xFFFFu) << 16);       // tensor_dim0[15:0]
  g1[2] = (int)((nelem >> 16) | (1u << 16));    // tensor_dim0[31:16] | tensor_dim1=1
  g1[3] = (int)(nelem << 16);                   // tile_dim0 (<=65535)
  g1[4] = 0;                                    // tile_dim1/2 unused
  g1[5] = (int)nelem;                           // tensor_dim0_stride[31:0]
  g1[6] = 0;
  g1[7] = 0;
  v4i z4 = {0, 0, 0, 0};
#if __has_include(<hip/amd_detail/amd_gfx1250_TDM.h>)
  v8i z8 = {0, 0, 0, 0, 0, 0, 0, 0};
  __builtin_amdgcn_tensor_load_to_lds(g0, g1, z4, z4, z8, 0);   // clang-23 arity
#else
  __builtin_amdgcn_tensor_load_to_lds(g0, g1, z4, z4, 0);       // ROCm 7.2 arity
#endif
#else
  (void)lds_off; (void)gsrc; (void)nelem;
#endif
}

static __device__ inline void tdm_wait() {
#if __has_builtin(__builtin_amdgcn_s_wait_tensorcnt)
  __builtin_amdgcn_s_wait_tensorcnt(0);
#endif
}

// Stage nelem bf16 from ws into LDS at byte offset lds_off. TDM when present,
// cooperative copy fallback otherwise. Caller must __syncthreads() after.
static __device__ inline void stage_lds(char* smem, u32 lds_off, const bf16* src,
                                        u32 nelem, int tid) {
#if HAVE_TDM
  if (tid < 32) tdm_load_1d(lds_off, src, nelem);
  tdm_wait();
#else
  bf16* dst = (bf16*)(smem + lds_off);
  for (u32 i = tid; i < nelem; i += 256) dst[i] = src[i];
#endif
}

// ---- 16-lane row reductions via v_permlane16 ------------------------------
constexpr unsigned xor_tab(int base, int d) {
  unsigned v = 0;
  for (int i = 0; i < 8; ++i) v |= (unsigned)(((base + i) ^ d) & 15) << (4 * i);
  return v;
}
template <int D>
static __device__ inline float row_xor(float x) {
#if __has_builtin(__builtin_amdgcn_permlane16)
  constexpr unsigned lo = xor_tab(0, D), hi = xor_tab(8, D);
  int xi = __float_as_int(x);
  int r = __builtin_amdgcn_permlane16(xi, xi, (int)lo, (int)hi, false, false);
  return __int_as_float(r);
#else
  return __shfl_xor(x, D);
#endif
}
static __device__ inline float red_sum16(float x) {
  x += row_xor<1>(x); x += row_xor<2>(x); x += row_xor<4>(x); x += row_xor<8>(x);
  return x;
}
static __device__ inline float red_max16(float x) {
  x = fmaxf(x, row_xor<1>(x)); x = fmaxf(x, row_xor<2>(x));
  x = fmaxf(x, row_xor<4>(x)); x = fmaxf(x, row_xor<8>(x));
  return x;
}
static __device__ inline float fast_rcp(float x) {
#if __has_builtin(__builtin_amdgcn_rcpf)
  return __builtin_amdgcn_rcpf(x);
#else
  return 1.f / x;
#endif
}

// ---- WMMA helpers ---------------------------------------------------------
static __device__ inline v8f wmma_bf16(v16bf a, v16bf b, v8f c) {
  return __builtin_amdgcn_wmma_f32_16x16x32_bf16(
      false, a, false, b, (short)0, c, false, false);
}

// A fragment (16x32) from f32 row-major LDS; pad k>=kmax with 0.
static __device__ inline v16bf frag_a_f32(const float* s, int ld, int kmax, float scale) {
  int lane = threadIdx.x & 31, half = lane >> 4, m = lane & 15, kb = half * 16;
  v16bf a;
#pragma unroll
  for (int i = 0; i < 8; ++i) {
    int k = kb + 2 * i;
    float x0 = (k     < kmax) ? s[m * ld + k]     * scale : 0.f;
    float x1 = (k + 1 < kmax) ? s[m * ld + k + 1] * scale : 0.f;
    a[2 * i]     = (bf16)x0;
    a[2 * i + 1] = (bf16)x1;
  }
  return a;
}

// B fragment (32x16) from bf16 row-major LDS [k*ld + n].
static __device__ inline v16bf frag_b_bf16(const bf16* s, int ld) {
  int lane = threadIdx.x & 31, n = lane & 15, kb = (lane >> 4) * 16;
  v16bf b;
#pragma unroll
  for (int j = 0; j < 8; ++j) {
    b[2 * j]     = s[(kb + 2 * j)     * ld + n];
    b[2 * j + 1] = s[(kb + 2 * j + 1) * ld + n];
  }
  return b;
}

// Transposed B fragment: B[k][n] = s[n*ld + k]; pad k>=kmax (for Q*K^T).
static __device__ inline v16bf frag_bT_bf16(const bf16* s, int ld, int kmax) {
  int lane = threadIdx.x & 31, n = lane & 15, kb = (lane >> 4) * 16;
  v16bf b;
#pragma unroll
  for (int j = 0; j < 8; ++j) {
    int k0 = kb + 2 * j, k1 = k0 + 1;
    b[2 * j]     = (k0 < kmax) ? s[n * ld + k0] : (bf16)0.f;
    b[2 * j + 1] = (k1 < kmax) ? s[n * ld + k1] : (bf16)0.f;
  }
  return b;
}

static __device__ inline void fourier17(float v, float* o) {
  o[0] = v;
  float f = 3.14159265358979323846f;
#pragma unroll
  for (int j = 0; j < NF_; ++j) {
    float a = v * f;
    o[1 + j] = __sinf(a);
    o[9 + j] = __cosf(a);
    f *= 2.f;
  }
}

// ---------------------------------------------------------------------------
// Kernel A: convert reused weights to bf16 in workspace (one contiguous
// region per future TDM transfer).
// ---------------------------------------------------------------------------
__global__ void wno_pack_kernel(DevP P, void* wsraw) {
  bf16* ws = (bf16*)wsraw;
  int gid = blockIdx.x * 256 + threadIdx.x;
  int gsz = gridDim.x * 256;
  for (int l = 0; l < NL_; ++l) {
    const LayerP L = P.L[l];
    bf16* dst = ws + LAY_OFF + l * LAY_STRIDE;
    for (int i = gid; i < HD_ * HD_; i += gsz) dst[i]                 = (bf16)L.Wq[i];
    for (int i = gid; i < HD_ * HD_; i += gsz) dst[HD_ * HD_ + i]     = (bf16)L.Wo[i];
    for (int i = gid; i < HD_ * FF_; i += gsz) dst[2 * HD_ * HD_ + i] = (bf16)L.W1[i];
    for (int i = gid; i < HD_ * FF_; i += gsz)
      dst[2 * HD_ * HD_ + HD_ * FF_ + i] = (bf16)L.W2[i];
  }
  for (int i = gid; i < HD_ * HD_; i += gsz) ws[HEAD_OFF + i] = (bf16)P.headW1[i];
}

// ---------------------------------------------------------------------------
// Kernel B: seg features -> seg MLP -> per-layer K/V (bf16) into workspace.
// ---------------------------------------------------------------------------
__global__ void wno_seg_kv_kernel(DevP P, void* wsraw) {
  bf16* ws = (bf16*)wsraw;
  int t = threadIdx.x, b = t >> 5, k = t & 31;
  float xl = P.xs[b * (KSEG + 1) + k], xr = P.xs[b * (KSEG + 1) + k + 1];
  float kk = P.ks[b * KSEG + k];
  float feats[6];
  feats[0] = xl; feats[1] = xr; feats[2] = kk;
  feats[3] = kk * (1.f - kk);
  feats[4] = 1.f - 2.f * kk;
  float mass = 0.f;
  for (int j = 0; j <= k; ++j)
    mass += P.ks[b * KSEG + j] * (P.xs[b * (KSEG + 1) + j + 1] - P.xs[b * (KSEG + 1) + j]);
  feats[5] = mass;

  float si[6 * FDIM_];
#pragma unroll
  for (int f = 0; f < 6; ++f) fourier17(feats[f], &si[f * FDIM_]);

  float h[HD_];
  for (int o = 0; o < HD_; ++o) {
    float a = P.segB1[o];
    for (int i = 0; i < 6 * FDIM_; ++i) a += si[i] * P.segW1[i * HD_ + o];
    h[o] = fmaxf(a, 0.f);
  }
  float e[HD_];
  for (int o = 0; o < HD_; ++o) {
    float a = P.segB2[o];
    for (int i = 0; i < HD_; ++i) a += h[i] * P.segW2[i * HD_ + o];
    e[o] = a;
  }
  for (int l = 0; l < NL_; ++l) {
    const LayerP& L = P.L[l];
    bf16* kvb = ws + (size_t)(l * B_ + b) * KV_PER_LB;   // [Kh | V]
    for (int o = 0; o < HD_; ++o) {
      float ak = L.bk[o], av = L.bv[o];
      for (int i = 0; i < HD_; ++i) {
        ak += e[i] * L.Wk[i * HD_ + o];
        av += e[i] * L.Wv[i * HD_ + o];
      }
      kvb[k * HD_ + o]                = (bf16)ak;
      kvb[KSEG * HD_ + k * HD_ + o]   = (bf16)av;
    }
  }
}

// ---------------------------------------------------------------------------
// Main fused kernel: one workgroup per (b, t). 256 threads = 8 waves,
// each wave owns 2 M-tiles of 16 query rows.
// ---------------------------------------------------------------------------
__global__ void wno_main_kernel(DevP P, const void* wsraw, float* outD, float* outB) {
  extern __shared__ char smem[];
  float* sX    = (float*)(smem + OFF_SX);
  float* sBias = (float*)(smem + OFF_BIAS);
  bf16*  sW    = (bf16*)(smem + OFF_W);
  bf16*  sKh   = (bf16*)(smem + OFF_KV);
  bf16*  sV    = sKh + KSEG * HD_;
  float* sScrA = (float*)(smem + OFF_SCR);
  float* sScr2A= (float*)(smem + OFF_SCR2);
  float* sXc   = (float*)(smem + OFF_XC);
  float* sXm   = (float*)(smem + OFF_KA);
  float* sSp   = sXm + KSEG;
  float* sMk   = sSp + KSEG;

  const bf16* ws = (const bf16*)wsraw;
  int tid = threadIdx.x, lane = tid & 31, wave = tid >> 5;
  int half = lane >> 4, l16 = lane & 15;
  int bn = blockIdx.x, b = bn >> 8, ti = bn & (NT_ - 1);

  float* myScr  = sScrA  + wave * 16 * 64;
  float* myScr2 = sScr2A + wave * 16 * 32;

  float tval   = P.tco[(size_t)(b * NT_ + ti) * NX_];
  float bscale = P.bias_scale[0];

  if (tid < NX_)  sXc[tid] = P.xco[(size_t)(b * NT_ + ti) * NX_ + tid];
  if (tid < KSEG) {
    int k = tid;
    float xl = P.xs[b * (KSEG + 1) + k], xr = P.xs[b * (KSEG + 1) + k + 1];
    float kk = P.ks[b * KSEG + k];
    sXm[k] = 0.5f * (xl + xr);
    sSp[k] = 1.f - 2.f * kk;
    sMk[k] = (P.pm[b * KSEG + k] == 0) ? -1.0e9f : 0.f;
  }
  __syncthreads();

  for (int idx = tid; idx < NX_ * KSEG; idx += 256) {
    int n = idx >> 5, k = idx & 31;
    float cb = -bscale * fabsf(sXc[n] - (sXm[k] + sSp[k] * tval));
    outB[((size_t)(b * NT_ + ti) * NX_ + n) * KSEG + k] = cb;
    sBias[n * KSEG + k] = cb + sMk[k];
  }

  // q init: one row per thread (34 -> 64 -> 64 MLP)
  {
    int n = tid;
    float qf[2 * FDIM_];
    fourier17(tval, qf);
    fourier17(sXc[n], qf + FDIM_);
    float h[HD_];
    for (int o = 0; o < HD_; ++o) {
      float a = P.qryB1[o];
      for (int i = 0; i < 2 * FDIM_; ++i) a += qf[i] * P.qryW1[i * HD_ + o];
      h[o] = fmaxf(a, 0.f);
    }
    for (int o = 0; o < HD_; ++o) {
      float a = P.qryB2[o];
      for (int i = 0; i < HD_; ++i) a += h[i] * P.qryW2[i * HD_ + o];
      sX[n * HD_ + o] = a;
    }
  }
  __syncthreads();

  const float kEps = 1e-5f;

  for (int l = 0; l < NL_; ++l) {
    const LayerP L = P.L[l];
    const bf16* wsLay = ws + LAY_OFF + l * LAY_STRIDE;
    __builtin_prefetch(wsLay, 0, 0);

    // ---- stage [Wq|Wo] and [Kh|V] via TDM ------------------------------
    bf16* sWq = sW;
    bf16* sWo = sW + HD_ * HD_;
#if HAVE_TDM
    if (tid < 32) {
      tdm_load_1d(OFF_W,  wsLay, 2 * HD_ * HD_);
      tdm_load_1d(OFF_KV, ws + (size_t)(l * B_ + b) * KV_PER_LB, KV_PER_LB);
    }
    tdm_wait();
#else
    stage_lds(smem, OFF_W,  wsLay, 2 * HD_ * HD_, tid);
    stage_lds(smem, OFF_KV, ws + (size_t)(l * B_ + b) * KV_PER_LB, KV_PER_LB, tid);
#endif
    __syncthreads();

    // ---- attention (row-private per wave) ------------------------------
    for (int it = 0; it < 2; ++it) {
      int r0 = (wave + 8 * it) * 16;
      v16bf aX0 = frag_a_f32(sX + r0 * HD_,      HD_, 32, 1.f);
      v16bf aX1 = frag_a_f32(sX + r0 * HD_ + 32, HD_, 32, 1.f);

      v8f qa[4];
#pragma unroll
      for (int nt = 0; nt < 4; ++nt) {
        v8f c = {};
        c = wmma_bf16(aX0, frag_b_bf16(sWq + nt * 16, HD_), c);
        c = wmma_bf16(aX1, frag_b_bf16(sWq + 32 * HD_ + nt * 16, HD_), c);
        qa[nt] = c;
      }
#pragma unroll
      for (int nt = 0; nt < 4; ++nt) {
        float bqv = L.bq[nt * 16 + l16];
#pragma unroll
        for (int r = 0; r < 8; ++r) {
          int m = r + 8 * half;
          myScr[m * 64 + nt * 16 + l16] = qa[nt][r] + bqv;
        }
      }

      v8f ctx[NH_];
#pragma unroll
      for (int h = 0; h < NH_; ++h) {
        v16bf aQ = frag_a_f32(myScr + 16 * h, 64, DH_, 0.25f);
        v8f s0, s1;
#pragma unroll
        for (int r = 0; r < 8; ++r) {
          int m = r + 8 * half;
          s0[r] = sBias[(r0 + m) * KSEG + l16];
          s1[r] = sBias[(r0 + m) * KSEG + 16 + l16];
        }
        s0 = wmma_bf16(aQ, frag_bT_bf16(sKh + 16 * h,            HD_, DH_), s0);
        s1 = wmma_bf16(aQ, frag_bT_bf16(sKh + 16 * HD_ + 16 * h, HD_, DH_), s1);

#pragma unroll
        for (int r = 0; r < 8; ++r) {
          float x0 = s0[r], x1 = s1[r];
          float mx = red_max16(fmaxf(x0, x1));
          float p0 = __expf(x0 - mx), p1 = __expf(x1 - mx);
          float inv = fast_rcp(red_sum16(p0 + p1));
          int m = r + 8 * half;
          myScr2[m * 32 + l16]      = p0 * inv;
          myScr2[m * 32 + 16 + l16] = p1 * inv;
        }
        v16bf aP = frag_a_f32(myScr2, 32, 32, 1.f);
        v8f z = {};
        ctx[h] = wmma_bf16(aP, frag_b_bf16(sV + 16 * h, HD_), z);
      }

#pragma unroll
      for (int h = 0; h < NH_; ++h)
#pragma unroll
        for (int r = 0; r < 8; ++r) {
          int m = r + 8 * half;
          myScr[m * 64 + 16 * h + l16] = ctx[h][r];
        }

      v16bf aC0 = frag_a_f32(myScr,      64, 32, 1.f);
      v16bf aC1 = frag_a_f32(myScr + 32, 64, 32, 1.f);
      v8f oa[4];
#pragma unroll
      for (int nt = 0; nt < 4; ++nt) {
        v8f c = {};
        c = wmma_bf16(aC0, frag_b_bf16(sWo + nt * 16, HD_), c);
        c = wmma_bf16(aC1, frag_b_bf16(sWo + 32 * HD_ + nt * 16, HD_), c);
        oa[nt] = c;
      }

#pragma unroll
      for (int r = 0; r < 8; ++r) {
        int m = r + 8 * half, gr = r0 + m;
        float v[4], s1_ = 0.f, s2_ = 0.f;
#pragma unroll
        for (int nt = 0; nt < 4; ++nt) {
          int n = nt * 16 + l16;
          float x = sX[gr * HD_ + n] + oa[nt][r] + L.bo[n];
          v[nt] = x; s1_ += x; s2_ += x * x;
        }
        s1_ = red_sum16(s1_);
        s2_ = red_sum16(s2_);
        float mu = s1_ * (1.f / HD_);
        float var = s2_ * (1.f / HD_) - mu * mu;
        float rs = rsqrtf(var + kEps);
#pragma unroll
        for (int nt = 0; nt < 4; ++nt) {
          int n = nt * 16 + l16;
          sX[gr * HD_ + n] = (v[nt] - mu) * rs * L.ln1g[n] + L.ln1b[n];
        }
      }
    }
    __syncthreads();

    // ---- stage [W1|W2] via TDM -----------------------------------------
    bf16* sW1 = sW;
    bf16* sW2 = sW + HD_ * FF_;
#if HAVE_TDM
    if (tid < 32) tdm_load_1d(OFF_W, wsLay + 2 * HD_ * HD_, 2 * HD_ * FF_);
    tdm_wait();
#else
    stage_lds(smem, OFF_W, wsLay + 2 * HD_ * HD_, 2 * HD_ * FF_, tid);
#endif
    __syncthreads();

    // ---- FFN: out = relu(X W1 + b1) W2, blocked over 32 hidden cols ----
    for (int it = 0; it < 2; ++it) {
      int r0 = (wave + 8 * it) * 16;
      v16bf aX0 = frag_a_f32(sX + r0 * HD_,      HD_, 32, 1.f);
      v16bf aX1 = frag_a_f32(sX + r0 * HD_ + 32, HD_, 32, 1.f);
      v8f oacc[4];
#pragma unroll
      for (int nt = 0; nt < 4; ++nt) { v8f z = {}; oacc[nt] = z; }

      for (int jp = 0; jp < 8; ++jp) {
        v8f t0 = {}, t1 = {};
        t0 = wmma_bf16(aX0, frag_b_bf16(sW1 + jp * 32,                 FF_), t0);
        t0 = wmma_bf16(aX1, frag_b_bf16(sW1 + 32 * FF_ + jp * 32,      FF_), t0);
        t1 = wmma_bf16(aX0, frag_b_bf16(sW1 + jp * 32 + 16,            FF_), t1);
        t1 = wmma_bf16(aX1, frag_b_bf16(sW1 + 32 * FF_ + jp * 32 + 16, FF_), t1);
        float bv0 = L.b1[jp * 32 + l16], bv1 = L.b1[jp * 32 + 16 + l16];
#pragma unroll
        for (int r = 0; r < 8; ++r) {
          int m = r + 8 * half;
          myScr2[m * 32 + l16]      = fmaxf(t0[r] + bv0, 0.f);
          myScr2[m * 32 + 16 + l16] = fmaxf(t1[r] + bv1, 0.f);
        }
        v16bf aH = frag_a_f32(myScr2, 32, 32, 1.f);
#pragma unroll
        for (int nt = 0; nt < 4; ++nt)
          oacc[nt] = wmma_bf16(aH, frag_b_bf16(sW2 + (jp * 32) * HD_ + nt * 16, HD_), oacc[nt]);
      }

#pragma unroll
      for (int r = 0; r < 8; ++r) {
        int m = r + 8 * half, gr = r0 + m;
        float v[4], s1_ = 0.f, s2_ = 0.f;
#pragma unroll
        for (int nt = 0; nt < 4; ++nt) {
          int n = nt * 16 + l16;
          float x = sX[gr * HD_ + n] + oacc[nt][r] + L.b2[n];
          v[nt] = x; s1_ += x; s2_ += x * x;
        }
        s1_ = red_sum16(s1_);
        s2_ = red_sum16(s2_);
        float mu = s1_ * (1.f / HD_);
        float var = s2_ * (1.f / HD_) - mu * mu;
        float rs = rsqrtf(var + kEps);
#pragma unroll
        for (int nt = 0; nt < 4; ++nt) {
          int n = nt * 16 + l16;
          sX[gr * HD_ + n] = (v[nt] - mu) * rs * L.ln2g[n] + L.ln2b[n];
        }
      }
    }
    __syncthreads();
  }

  // ---- head: density = clip(relu(X hW1 + hb1) . hW2 + hb2, 0, 1) --------
#if HAVE_TDM
  if (tid < 32) tdm_load_1d(OFF_W, ws + HEAD_OFF, HD_ * HD_);
  tdm_wait();
#else
  stage_lds(smem, OFF_W, ws + HEAD_OFF, HD_ * HD_, tid);
#endif
  __syncthreads();

  for (int it = 0; it < 2; ++it) {
    int r0 = (wave + 8 * it) * 16;
    v16bf a0 = frag_a_f32(sX + r0 * HD_,      HD_, 32, 1.f);
    v16bf a1 = frag_a_f32(sX + r0 * HD_ + 32, HD_, 32, 1.f);
    v8f acc[4];
#pragma unroll
    for (int nt = 0; nt < 4; ++nt) {
      v8f c = {};
      c = wmma_bf16(a0, frag_b_bf16(sW + nt * 16, HD_), c);
      c = wmma_bf16(a1, frag_b_bf16(sW + 32 * HD_ + nt * 16, HD_), c);
      acc[nt] = c;
    }
#pragma unroll
    for (int r = 0; r < 8; ++r) {
      int m = r + 8 * half;
      float p = 0.f;
#pragma unroll
      for (int nt = 0; nt < 4; ++nt) {
        int n = nt * 16 + l16;
        p += fmaxf(acc[nt][r] + P.headB1[n], 0.f) * P.headW2[n];
      }
      p = red_sum16(p);
      if (l16 == 0) {
        float d = p + P.headB2[0];
        d = fminf(fmaxf(d, 0.f), 1.f);
        outD[(size_t)(b * NT_ + ti) * NX_ + r0 + m] = d;
      }
    }
  }
}

// ---------------------------------------------------------------------------
// Host launcher. Input order: top-level setup_inputs() insertion order, then
// params flattened jax-pytree style (dict keys sorted; 'layers' list in place).
// ---------------------------------------------------------------------------
extern "C" void kernel_launch(void* const* d_in, const int* in_sizes, int n_in,
                              void* d_out, int out_size, void* d_ws, size_t ws_size,
                              hipStream_t stream) {
  (void)in_sizes; (void)n_in; (void)out_size; (void)ws_size;
  DevP P;
  P.xs  = (const float*)d_in[0];  P.ks  = (const float*)d_in[1];
  P.pm  = (const int*)  d_in[2];
  P.tco = (const float*)d_in[3];  P.xco = (const float*)d_in[4];
  P.bias_scale = (const float*)d_in[5];
  P.headW1 = (const float*)d_in[6];  P.headW2 = (const float*)d_in[7];
  P.headB1 = (const float*)d_in[8];  P.headB2 = (const float*)d_in[9];
  for (int l = 0; l < NL_; ++l) {
    const int base = 10 + 16 * l;
    LayerP& L = P.L[l];
    L.W1 = (const float*)d_in[base + 0];  L.W2 = (const float*)d_in[base + 1];
    L.Wk = (const float*)d_in[base + 2];  L.Wo = (const float*)d_in[base + 3];
    L.Wq = (const float*)d_in[base + 4];  L.Wv = (const float*)d_in[base + 5];
    L.b1 = (const float*)d_in[base + 6];  L.b2 = (const float*)d_in[base + 7];
    L.bk = (const float*)d_in[base + 8];  L.bo = (const float*)d_in[base + 9];
    L.bq = (const float*)d_in[base + 10]; L.bv = (const float*)d_in[base + 11];
    L.ln1b = (const float*)d_in[base + 12]; L.ln1g = (const float*)d_in[base + 13];
    L.ln2b = (const float*)d_in[base + 14]; L.ln2g = (const float*)d_in[base + 15];
  }
  P.qryW1 = (const float*)d_in[42]; P.qryW2 = (const float*)d_in[43];
  P.qryB1 = (const float*)d_in[44]; P.qryB2 = (const float*)d_in[45];
  P.segW1 = (const float*)d_in[46]; P.segW2 = (const float*)d_in[47];
  P.segB1 = (const float*)d_in[48]; P.segB2 = (const float*)d_in[49];

  float* outD = (float*)d_out;
  float* outB = outD + (size_t)B_ * NT_ * NX_;

  (void)hipFuncSetAttribute(reinterpret_cast<const void*>(wno_main_kernel),
                            hipFuncAttributeMaxDynamicSharedMemorySize, SMEM_BYTES);

  wno_pack_kernel<<<64, 256, 0, stream>>>(P, d_ws);
  wno_seg_kv_kernel<<<1, 256, 0, stream>>>(P, d_ws);
  wno_main_kernel<<<BN_, 256, SMEM_BYTES, stream>>>(P, d_ws, outD, outB);
}
